// SSBlock_11063835754552
// MI455X (gfx1250) — compile-verified, matrix operand order checked by reading, not run on hardware
//
#include <hip/hip_runtime.h>

// ---------------- CDNA5 WMMA types ----------------
typedef __attribute__((ext_vector_type(16))) __bf16 v16bf;
typedef __attribute__((ext_vector_type(8)))  float  v8f;

#define B_   8
#define L_   2048
#define M_   16384   // B_*L_
#define DM_  64
#define DP_  64
#define DI_  128
#define NS_  16
#define NX_  48      // padded x_dbl width (valid 36)

__device__ __forceinline__ v8f wmma_bf16(v16bf a, v16bf b, v8f c) {
  return __builtin_amdgcn_wmma_f32_16x16x32_bf16(false, a, false, b, (short)0, c, false, false);
}

// Load a 16x32 fragment (A-layout; also used for B since weights are (N,K) row-major).
// lane 0..15 -> row = row0+lane, elements e: K = (e>>3)*16 + (lane>=16 ? 8:0) + (e&7)
__device__ __forceinline__ v16bf ld_frag(const float* __restrict__ base, int row0,
                                         long rowStride, int k0, int elemStride,
                                         int lane, float scale) {
  int r = row0 + (lane & 15);
  int g = (lane >> 4) & 1;
  const float* p = base + (long)r * rowStride + (long)k0 * elemStride;
  v16bf f;
#pragma unroll
  for (int e = 0; e < 16; ++e) {
    int k = ((e >> 3) << 4) + (g << 3) + (e & 7);
    f[e] = (__bf16)(scale * p[(long)k * elemStride]);
  }
  return f;
}

__device__ __forceinline__ v16bf ld_frag_guard(const float* __restrict__ base, int row0,
                                               int rowMax, long rowStride, int k0, int lane) {
  int r = row0 + (lane & 15);
  int g = (lane >> 4) & 1;
  v16bf f;
  if (r < rowMax) {
    const float* p = base + (long)r * rowStride + k0;
#pragma unroll
    for (int e = 0; e < 16; ++e) {
      int k = ((e >> 3) << 4) + (g << 3) + (e & 7);
      f[e] = (__bf16)p[k];
    }
  } else {
#pragma unroll
    for (int e = 0; e < 16; ++e) f[e] = (__bf16)0.0f;
  }
  return f;
}

__device__ __forceinline__ float4 comp4(float4 l, float4 r) {
  // scan combine, l applied first: A = A_l*A_r (complex), B = B_l*A_r + B_r
  float4 o;
  o.x = l.x * r.x - l.y * r.y;
  o.y = l.x * r.y + l.y * r.x;
  o.z = l.z * r.x - l.w * r.y + r.z;
  o.w = l.z * r.y + l.w * r.x + r.w;
  return o;
}

// ---------------- K1: S5 Bu GEMM (WMMA) + ZOH discretization -> (a,b) complex ----------------
__global__ void k1_s5_ab(const float* __restrict__ u, const float* __restrict__ tint,
                         const float* __restrict__ lam_re, const float* __restrict__ lam_im,
                         const float* __restrict__ Bre, const float* __restrict__ Bim,
                         const float* __restrict__ logstep, float4* __restrict__ ab) {
  int lane = threadIdx.x & 31;
  int w    = threadIdx.x >> 5;        // ntile 0..3 over P=64
  int m0 = blockIdx.x * 16, n0 = w * 16;
  v8f accr = {}; v8f acci = {};
  for (int k0 = 0; k0 < DM_; k0 += 32) {
    v16bf a  = ld_frag(u,   m0, DM_, k0, 1, lane, 1.f);
    v16bf br = ld_frag(Bre, n0, DM_, k0, 1, lane, 1.f);
    v16bf bi = ld_frag(Bim, n0, DM_, k0, 1, lane, 1.f);
    accr = wmma_bf16(a, br, accr);
    acci = wmma_bf16(a, bi, acci);
  }
  int n  = n0 + (lane & 15);
  int mb = (lane >> 4) * 8;
  float ls = __expf(logstep[n]);
  float lr = lam_re[n], li = lam_im[n];
  float den = lr * lr + li * li;
#pragma unroll
  for (int v = 0; v < 8; ++v) {
    long m = m0 + mb + v;
    float dtv = tint[m] * ls;
    float er = __expf(lr * dtv);
    float s, c; __sincosf(li * dtv, &s, &c);
    float ar = er * c, ai = er * s;           // Lambda_bar
    float nr = ar - 1.f, ni = ai;             // (Lambda_bar - 1)
    float qr = (nr * lr + ni * li) / den;     // /Lambda  (complex)
    float qi = (ni * lr - nr * li) / den;
    float br_ = qr * accr[v] - qi * acci[v];  // * Bu (complex)
    float bi_ = qr * acci[v] + qi * accr[v];
    ab[m * DP_ + n] = make_float4(ar, ai, br_, bi_);
  }
}

// ---------------- K2: S5 complex chunked scan over L (one block per (b,p)) ----------------
__global__ void k2_s5_scan(const float4* __restrict__ ab, float2* __restrict__ xs) {
  __shared__ float4 s[256];
  int bb = blockIdx.x >> 6;   // /64
  int p  = blockIdx.x & 63;
  int t  = threadIdx.x;       // 256 threads, 8 steps each
  long base = (long)bb * L_;
  float4 e[8];
  float4 agg = make_float4(1.f, 0.f, 0.f, 0.f);
#pragma unroll
  for (int j = 0; j < 8; ++j) {
    float4 v = ab[(base + t * 8 + j) * DP_ + p];
    e[j] = v;
    agg = comp4(agg, v);
  }
  s[t] = agg;
  __syncthreads();
  for (int off = 1; off < 256; off <<= 1) {
    float4 left = (t >= off) ? s[t - off] : make_float4(1.f, 0.f, 0.f, 0.f);
    __syncthreads();
    s[t] = comp4(left, s[t]);
    __syncthreads();
  }
  float4 pre = (t > 0) ? s[t - 1] : make_float4(1.f, 0.f, 0.f, 0.f);
  float sr = pre.z, si = pre.w;
#pragma unroll
  for (int j = 0; j < 8; ++j) {
    float4 v = e[j];
    float nr = sr * v.x - si * v.y + v.z;     // x = a*x + b (complex)
    float ni = sr * v.y + si * v.x + v.w;
    sr = nr; si = ni;
    xs[(base + t * 8 + j) * DP_ + p] = make_float2(sr, si);
  }
}

// ---------------- K3: y = 2*Re(xs @ C^T) + (D+1)*u, fused LayerNorm -> x1 ----------------
__global__ void k3_s5out_ln(const float* __restrict__ xsf, const float* __restrict__ Cre,
                            const float* __restrict__ Cim, const float* __restrict__ Ds5,
                            const float* __restrict__ u, const float* __restrict__ g,
                            const float* __restrict__ bta, float* __restrict__ x1) {
  __shared__ float red[2][16][4];
  __shared__ float mu_s[16], rs_s[16];
  int lane = threadIdx.x & 31;
  int w    = threadIdx.x >> 5;
  int m0 = blockIdx.x * 16, n0 = w * 16;
  v8f acc = {};
  for (int k0 = 0; k0 < DP_; k0 += 32) {
    v16bf ar = ld_frag(xsf,     m0, 2 * DP_, k0, 2, lane,  1.f); // xs_re
    v16bf ai = ld_frag(xsf + 1, m0, 2 * DP_, k0, 2, lane, -1.f); // -xs_im
    v16bf br = ld_frag(Cre, n0, DP_, k0, 1, lane, 1.f);
    v16bf bi = ld_frag(Cim, n0, DP_, k0, 1, lane, 1.f);
    acc = wmma_bf16(ar, br, acc);   //  xs_re . C_re
    acc = wmma_bf16(ai, bi, acc);   // -xs_im . C_im
  }
  int n  = n0 + (lane & 15);
  int mb = (lane >> 4) * 8;
  float vals[8];
#pragma unroll
  for (int v = 0; v < 8; ++v) {
    long m = m0 + mb + v;
    vals[v] = 2.f * acc[v] + (Ds5[n] + 1.f) * u[m * DM_ + n]; // + residual input
  }
#pragma unroll
  for (int v = 0; v < 8; ++v) {
    float s1 = vals[v], s2 = vals[v] * vals[v];
#pragma unroll
    for (int msk = 1; msk < 16; msk <<= 1) {
      s1 += __shfl_xor(s1, msk, 32);
      s2 += __shfl_xor(s2, msk, 32);
    }
    if ((lane & 15) == 0) { red[0][mb + v][w] = s1; red[1][mb + v][w] = s2; }
  }
  __syncthreads();
  if (threadIdx.x < 16) {
    int r = threadIdx.x;
    float s1 = red[0][r][0] + red[0][r][1] + red[0][r][2] + red[0][r][3];
    float s2 = red[1][r][0] + red[1][r][1] + red[1][r][2] + red[1][r][3];
    float mu = s1 / 64.f;
    mu_s[r] = mu;
    rs_s[r] = rsqrtf(s2 / 64.f - mu * mu + 1e-12f);
  }
  __syncthreads();
#pragma unroll
  for (int v = 0; v < 8; ++v) {
    int r = mb + v;
    long m = m0 + r;
    x1[m * DM_ + n] = (vals[v] - mu_s[r]) * rs_s[r] * g[n] + bta[n];
  }
}

// ---------------- K4: generic per-wave 16x16-tile GEMM  O[M,N] = A[M,K] @ W[N,K]^T ----------------
__global__ void k_gemm(const float* __restrict__ A, const float* __restrict__ W,
                       float* __restrict__ O, int Kd, int Nd) {
  int wid  = (int)(blockIdx.x * blockDim.x + threadIdx.x) >> 5;
  int lane = threadIdx.x & 31;
  int ntiles = Nd >> 4;
  int n0 = (wid % ntiles) * 16;
  int m0 = (wid / ntiles) * 16;
  v8f acc = {};
  for (int k0 = 0; k0 < Kd; k0 += 32) {
    v16bf a = ld_frag(A, m0, Kd, k0, 1, lane, 1.f);
    v16bf b = ld_frag(W, n0, Kd, k0, 1, lane, 1.f);
    acc = wmma_bf16(a, b, acc);
  }
  int n  = n0 + (lane & 15);
  int mb = (lane >> 4) * 8;
#pragma unroll
  for (int v = 0; v < 8; ++v) O[(long)(m0 + mb + v) * Nd + n] = acc[v];
}

// ---------------- K5: causal depthwise conv (width 4) + SiLU ----------------
__global__ void k5_conv(const float* __restrict__ xz, const float* __restrict__ cw,
                        const float* __restrict__ cb, float* __restrict__ xm) {
  long i = (long)blockIdx.x * blockDim.x + threadIdx.x;  // M_*DI_
  int  d = (int)(i & (DI_ - 1));
  long m = i >> 7;
  int  l = (int)(m & (L_ - 1));
  float s = cb[d];
#pragma unroll
  for (int j = 0; j < 4; ++j) {
    int lj = l - 3 + j;
    if (lj >= 0) s += xz[(m - 3 + j) * (2 * DI_) + d] * cw[d * 4 + j];
  }
  xm[i] = s / (1.f + __expf(-s));
}

// ---------------- K6: x_dbl = xm @ x_proj_w^T (N padded 36->48, guarded) ----------------
__global__ void k6_xdbl(const float* __restrict__ xm, const float* __restrict__ Wx,
                        float* __restrict__ xdbl) {
  int lane = threadIdx.x & 31;
  int w    = threadIdx.x >> 5;          // 3 waves -> 3 N tiles
  int m0 = blockIdx.x * 16, n0 = w * 16;
  v8f acc = {};
  for (int k0 = 0; k0 < DI_; k0 += 32) {
    v16bf a = ld_frag(xm, m0, DI_, k0, 1, lane, 1.f);
    v16bf b = ld_frag_guard(Wx, n0, 36, DI_, k0, lane);
    acc = wmma_bf16(a, b, acc);
  }
  int n  = n0 + (lane & 15);
  int mb = (lane >> 4) * 8;
#pragma unroll
  for (int v = 0; v < 8; ++v) xdbl[(long)(m0 + mb + v) * NX_ + n] = acc[v];
}

// ---------------- K7: dt = softplus(x_dbl[:, :4] @ dt_proj_w^T + b) ----------------
__global__ void k7_dt(const float* __restrict__ xdbl, const float* __restrict__ Wdt,
                      const float* __restrict__ bdt, float* __restrict__ dtb) {
  long i = (long)blockIdx.x * blockDim.x + threadIdx.x;  // M_*DI_
  int  d = (int)(i & (DI_ - 1));
  long m = i >> 7;
  float s = bdt[d];
#pragma unroll
  for (int r = 0; r < 4; ++r) s += xdbl[m * NX_ + r] * Wdt[d * 4 + r];
  dtb[i] = (s > 20.f) ? s : log1pf(__expf(s));
}

// ---------------- K8: fused selective scan + C-contraction + gate (block per (b,d)) ----------------
__global__ void k8_sscan(const float* __restrict__ dtb, const float* __restrict__ xdbl,
                         const float* __restrict__ xm, const float* __restrict__ xz,
                         const float* __restrict__ Alog, const float* __restrict__ Dm,
                         float* __restrict__ yb) {
  __shared__ float sA[256], sB[256];
  int bb = blockIdx.x >> 7;       // /128
  int d  = blockIdx.x & (DI_ - 1);
  int t = threadIdx.x, n = t & 15, c = t >> 4;     // state idx, time chunk
  float Adn = -__expf(Alog[d * NS_ + n]);
  long base = (long)bb * L_;
  int  l0 = c * 128;
  float a_run = 1.f, b_run = 0.f;
  for (int j = 0; j < 128; ++j) {
    long m = base + l0 + j;
    float dtv = dtb[m * DI_ + d];
    float a   = __expf(dtv * Adn);
    float bs  = dtv * xdbl[m * NX_ + 4 + n] * xm[m * DI_ + d];
    b_run = b_run * a + bs;
    a_run *= a;
  }
  sA[t] = a_run; sB[t] = b_run;
  __syncthreads();
  for (int off = 1; off < 16; off <<= 1) {
    float la = 1.f, lb = 0.f;
    if (c >= off) { la = sA[(c - off) * 16 + n]; lb = sB[(c - off) * 16 + n]; }
    __syncthreads();
    float ca = sA[t], cb = sB[t];
    sA[t] = la * ca; sB[t] = lb * ca + cb;
    __syncthreads();
  }
  float state = (c > 0) ? sB[(c - 1) * 16 + n] : 0.f;
  float Dd = Dm[d];
  for (int j = 0; j < 128; ++j) {
    long m = base + l0 + j;
    float dtv = dtb[m * DI_ + d];
    float xv  = xm[m * DI_ + d];
    float a   = __expf(dtv * Adn);
    float bs  = dtv * xdbl[m * NX_ + 4 + n] * xv;
    state = a * state + bs;
    float contrib = state * xdbl[m * NX_ + 20 + n];
#pragma unroll
    for (int msk = 1; msk < 16; msk <<= 1) contrib += __shfl_xor(contrib, msk, 32);
    if (n == 0) {
      float y = contrib + Dd * xv;
      float z = xz[m * (2 * DI_) + DI_ + d];
      y *= z / (1.f + __expf(-z));
      yb[m * DI_ + d] = y;
    }
  }
}

// ---------------- K9: out = y @ out_proj_w^T + x1, fused LayerNorm ----------------
__global__ void k9_out_ln(const float* __restrict__ yb, const float* __restrict__ Wo,
                          const float* __restrict__ x1, const float* __restrict__ g,
                          const float* __restrict__ bta, float* __restrict__ out) {
  __shared__ float red[2][16][4];
  __shared__ float mu_s[16], rs_s[16];
  int lane = threadIdx.x & 31;
  int w    = threadIdx.x >> 5;
  int m0 = blockIdx.x * 16, n0 = w * 16;
  v8f acc = {};
  for (int k0 = 0; k0 < DI_; k0 += 32) {
    v16bf a = ld_frag(yb, m0, DI_, k0, 1, lane, 1.f);
    v16bf b = ld_frag(Wo, n0, DI_, k0, 1, lane, 1.f);
    acc = wmma_bf16(a, b, acc);
  }
  int n  = n0 + (lane & 15);
  int mb = (lane >> 4) * 8;
  float vals[8];
#pragma unroll
  for (int v = 0; v < 8; ++v) {
    long m = m0 + mb + v;
    vals[v] = acc[v] + x1[m * DM_ + n];   // residual with normalized s5 output
  }
#pragma unroll
  for (int v = 0; v < 8; ++v) {
    float s1 = vals[v], s2 = vals[v] * vals[v];
#pragma unroll
    for (int msk = 1; msk < 16; msk <<= 1) {
      s1 += __shfl_xor(s1, msk, 32);
      s2 += __shfl_xor(s2, msk, 32);
    }
    if ((lane & 15) == 0) { red[0][mb + v][w] = s1; red[1][mb + v][w] = s2; }
  }
  __syncthreads();
  if (threadIdx.x < 16) {
    int r = threadIdx.x;
    float s1 = red[0][r][0] + red[0][r][1] + red[0][r][2] + red[0][r][3];
    float s2 = red[1][r][0] + red[1][r][1] + red[1][r][2] + red[1][r][3];
    float mu = s1 / 64.f;
    mu_s[r] = mu;
    rs_s[r] = rsqrtf(s2 / 64.f - mu * mu + 1e-12f);
  }
  __syncthreads();
#pragma unroll
  for (int v = 0; v < 8; ++v) {
    int r = mb + v;
    long m = m0 + r;
    out[m * DM_ + n] = (vals[v] - mu_s[r]) * rs_s[r] * g[n] + bta[n];
  }
}

// ---------------- K10: pass-through time_interval to output tuple ----------------
__global__ void k10_copy(const float* __restrict__ t, float* __restrict__ o) {
  int i = blockIdx.x * blockDim.x + threadIdx.x;
  if (i < M_) o[i] = t[i];
}

extern "C" void kernel_launch(void* const* d_in, const int* in_sizes, int n_in,
                              void* d_out, int out_size, void* d_ws, size_t ws_size,
                              hipStream_t stream) {
  (void)in_sizes; (void)n_in; (void)out_size; (void)ws_size;
  const float* input   = (const float*)d_in[0];
  const float* tint    = (const float*)d_in[1];
  const float* lam_re  = (const float*)d_in[2];
  const float* lam_im  = (const float*)d_in[3];
  const float* Bre     = (const float*)d_in[4];
  const float* Bim     = (const float*)d_in[5];
  const float* Cre     = (const float*)d_in[6];
  const float* Cim     = (const float*)d_in[7];
  const float* Ds5     = (const float*)d_in[8];
  const float* logstep = (const float*)d_in[9];
  const float* Win     = (const float*)d_in[10];
  const float* convw   = (const float*)d_in[11];
  const float* convb   = (const float*)d_in[12];
  const float* Wx      = (const float*)d_in[13];
  const float* Wdt     = (const float*)d_in[14];
  const float* bdt     = (const float*)d_in[15];
  const float* Alog    = (const float*)d_in[16];
  const float* Dm      = (const float*)d_in[17];
  const float* Wout    = (const float*)d_in[18];
  const float* lng     = (const float*)d_in[19];
  const float* lnb     = (const float*)d_in[20];
  float* out = (float*)d_out;

  // workspace layout (floats), with aliasing: xz reuses ab, yb reuses xs
  float* ws   = (float*)d_ws;
  float* ab   = ws;                    // 16384*64*4 = 4,194,304
  float* xz   = ws;                    // alias (ab dead after K2)
  float* xs   = ws + 4194304;          // 16384*64*2 = 2,097,152
  float* yb   = xs;                    // alias (xs dead after K3)
  float* x1   = ws + 6291456;          // 16384*64   = 1,048,576
  float* xm   = ws + 7340032;          // 16384*128  = 2,097,152
  float* xdbl = ws + 9437184;          // 16384*48   =   786,432
  float* dtb  = ws + 10223616;         // 16384*128  = 2,097,152

  k1_s5_ab  <<<M_ / 16, 128, 0, stream>>>(input, tint, lam_re, lam_im, Bre, Bim, logstep,
                                          (float4*)ab);
  k2_s5_scan<<<B_ * DP_, 256, 0, stream>>>((const float4*)ab, (float2*)xs);
  k3_s5out_ln<<<M_ / 16, 128, 0, stream>>>(xs, Cre, Cim, Ds5, input, lng, lnb, x1);
  // in_proj: M=16384, K=64, N=256 -> 16384 waves
  k_gemm    <<<2048, 256, 0, stream>>>(x1, Win, xz, DM_, 2 * DI_);
  k5_conv   <<<(M_ * DI_) / 256, 256, 0, stream>>>(xz, convw, convb, xm);
  k6_xdbl   <<<M_ / 16, 96, 0, stream>>>(xm, Wx, xdbl);
  k7_dt     <<<(M_ * DI_) / 256, 256, 0, stream>>>(xdbl, Wdt, bdt, dtb);
  k8_sscan  <<<B_ * DI_, 256, 0, stream>>>(dtb, xdbl, xm, xz, Alog, Dm, yb);
  k9_out_ln <<<M_ / 16, 128, 0, stream>>>(yb, Wout, x1, lng, lnb, out);
  k10_copy  <<<M_ / 256, 256, 0, stream>>>(tint, out + (long)M_ * DM_);
}
